// PrefixLMTransformer_35639638622862
// MI455X (gfx1250) — matrix operand order repaired and checked by reading, not compiled, not run on hardware
//
#include <hip/hip_runtime.h>
#include <hip/hip_bf16.h>
#include <stdint.h>

// ---------------------------------------------------------------------------
// PrefixLM transformer forward, MI455X (gfx1250), bf16 WMMA everywhere.
//   - GEMMs: C = A(bf16) @ W^T(f32->bf16) + bias, fused epilogues.
//     Software-pipelined staging: global->VGPR loads issued before the WMMA
//     burst, VGPR->LDS stores (and the forced s_wait_loadcnt) after it;
//     double-buffered LDS, one barrier per K-step; global_prefetch_b8 at
//     distance 2 to warm L2 for the 16x weight-tile reuse sweep.
//   - Attention: flash-style, fused softmax, wave32 shuffles, WMMA QK^T & PV.
// ---------------------------------------------------------------------------

constexpr int kV = 32000, kD = 1024, kL = 4, kH = 16, kS = 1024, kB = 2,
              kFF = 4096, kDH = 64, kPrefix = 256;
constexpr int kRows = kB * kS; // 2048 token rows

typedef __attribute__((ext_vector_type(16))) __bf16 bf16x16;
typedef __attribute__((ext_vector_type(8)))  float  f32x8;

union Frag16 {
  bf16x16 v;
  uint32_t u[8];
  __bf16 h[16];
};

static __device__ inline __bf16 f2bf(float f) {
  union { float f; uint32_t u; } a; a.f = f;
  uint32_t r = (a.u + 0x7FFFu + ((a.u >> 16) & 1u)) >> 16; // RNE
  union { uint16_t u; __bf16 h; } o; o.u = (uint16_t)r;
  return o.h;
}

static __device__ inline f32x8 wmma_bf16(const Frag16& a, const Frag16& b, f32x8 c) {
  return __builtin_amdgcn_wmma_f32_16x16x32_bf16(false, a.v, false, b.v,
                                                 (short)0, c, false, false);
}

// ---------------------------------------------------------------------------
// Embedding gather: x[row,:] = embedding[ids[row],:]
// ---------------------------------------------------------------------------
__global__ __launch_bounds__(256) void embed_gather(
    const int* __restrict__ ids, const float* __restrict__ emb,
    float* __restrict__ x) {
  const int row = blockIdx.x;
  const int tok = ids[row];
  const float* er = emb + (size_t)tok * kD;
  for (int i = threadIdx.x; i < kD; i += 256)
    x[(size_t)row * kD + i] = er[i];
}

// ---------------------------------------------------------------------------
// LayerNorm (f32 in) -> bf16 out. One block per token row.
// ---------------------------------------------------------------------------
__global__ __launch_bounds__(256) void layernorm_to_bf16(
    const float* __restrict__ x, const float* __restrict__ g,
    const float* __restrict__ bta, __bf16* __restrict__ out) {
  __shared__ float red[16];
  const int row = blockIdx.x;
  const float* xr = x + (size_t)row * kD;
  float s = 0.f, s2 = 0.f;
  for (int i = threadIdx.x; i < kD; i += 256) {
    const float v = xr[i];
    s += v; s2 += v * v;
  }
  for (int m = 1; m < 32; m <<= 1) {          // wave32 butterfly
    s  += __shfl_xor(s, m, 32);
    s2 += __shfl_xor(s2, m, 32);
  }
  const int wave = threadIdx.x >> 5, lane = threadIdx.x & 31;
  if (lane == 0) { red[wave] = s; red[8 + wave] = s2; }
  __syncthreads();
  s = 0.f; s2 = 0.f;
  for (int w = 0; w < 8; ++w) { s += red[w]; s2 += red[8 + w]; }
  const float mean = s * (1.0f / kD);
  const float var  = s2 * (1.0f / kD) - mean * mean;
  const float inv  = rsqrtf(var + 1e-5f);
  for (int i = threadIdx.x; i < kD; i += 256)
    out[(size_t)row * kD + i] = f2bf((xr[i] - mean) * inv * g[i] + bta[i]);
}

// ---------------------------------------------------------------------------
// Tiled WMMA GEMM: C[M,N] = A[M,K](bf16) @ W[N,K]^T + bias, fused epilogues.
//   epilogue 0: outF = acc+bias            (head logits, f32)
//   epilogue 1: outH = bf16(acc+bias)      (QKV)
//   epilogue 2: outH = bf16(gelu_erf(acc+bias))
//   epilogue 3: outF = residual + acc+bias (out-proj / FF2, residual stream)
// Block tile 128x128x32; 8 waves (2x4), each wave 64x32 via 4x2 WMMA tiles.
// ---------------------------------------------------------------------------
#define GBM 128
#define GBN 128
#define GBK 32

__global__ __launch_bounds__(256) void gemm_bf16_wmma(
    const __bf16* __restrict__ A, const float* __restrict__ W,
    const float* __restrict__ bias, const float* __restrict__ residual,
    float* __restrict__ outF, __bf16* __restrict__ outH,
    int M, int N, int K, int epilogue) {
  __shared__ __bf16 As[2][GBM][GBK];
  __shared__ __bf16 Bs[2][GBN][GBK];   // stored transposed: Bs[s][n][k]
  const int t = threadIdx.x;
  const int lane = t & 31;
  const int wave = t >> 5;
  const int waveM = wave >> 2;      // 0..1
  const int waveN = wave & 3;       // 0..3
  const int bm0 = blockIdx.y * GBM;
  const int bn0 = blockIdx.x * GBN;
  const int row16 = lane & 15;
  const int half = lane >> 4;
  // Per-thread stage coordinates: 1024 (row,16B-chunk) pairs over 256 threads.
  const int sr = t >> 1;            // row 0..127 (for i-th pass: +0, no — see e)
  (void)sr;

  f32x8 acc[4][2];
#pragma unroll
  for (int i = 0; i < 4; ++i)
#pragma unroll
    for (int j = 0; j < 2; ++j)
#pragma unroll
      for (int e = 0; e < 8; ++e) acc[i][j][e] = 0.0f;

  // Register staging: global -> VGPR (no wait at issue site).
  uint2  aregs[4];
  float4 wregs[4];
  auto gload = [&](int kb) {
    const int k0 = kb * GBK;
#pragma unroll
    for (int i = 0; i < 4; ++i) {
      const int e = t + i * 256;
      const int r = e >> 3;
      const int c = (e & 7) << 2;
      aregs[i] = *reinterpret_cast<const uint2*>(&A[(size_t)(bm0 + r) * K + k0 + c]);
      wregs[i] = *reinterpret_cast<const float4*>(&W[(size_t)(bn0 + r) * K + k0 + c]);
    }
  };
  // VGPR -> LDS (s_wait_loadcnt is forced here, after the WMMA burst).
  auto lstore = [&](int s) {
#pragma unroll
    for (int i = 0; i < 4; ++i) {
      const int e = t + i * 256;
      const int r = e >> 3;
      const int c = (e & 7) << 2;
      *reinterpret_cast<uint2*>(&As[s][r][c]) = aregs[i];
      __bf16* dst = &Bs[s][r][c];
      dst[0] = f2bf(wregs[i].x); dst[1] = f2bf(wregs[i].y);
      dst[2] = f2bf(wregs[i].z); dst[3] = f2bf(wregs[i].w);
    }
  };

  const int nk = K / GBK;
  gload(0);
  lstore(0);
  __syncthreads();

  for (int kb = 0; kb < nk; ++kb) {
    const int cur = kb & 1;
    // Issue next tile's global loads first; their wait lands after the WMMAs.
    if (kb + 1 < nk) gload(kb + 1);
    // Warm L2 for tile kb+2 (first HBM touch off the critical path).
    if (kb + 2 < nk) {
      const int k0p = (kb + 2) * GBK;
      const int pr = t >> 1;
      const int pc = (t & 1) * 16;
      __builtin_prefetch(&A[(size_t)(bm0 + pr) * K + k0p + pc], 0, 1);
      __builtin_prefetch(&W[(size_t)(bn0 + pr) * K + k0p + pc], 0, 1);
    }

    Frag16 af[4], bfr[2];
    // A frags: lane M=lane&15; halves K = half*8 + {0..7} and half*8+16+{0..7}
#pragma unroll
    for (int i = 0; i < 4; ++i) {
      const int r = waveM * 64 + i * 16 + row16;
      const uint32_t* p0 = reinterpret_cast<const uint32_t*>(&As[cur][r][half * 8]);
      const uint32_t* p1 = reinterpret_cast<const uint32_t*>(&As[cur][r][half * 8 + 16]);
#pragma unroll
      for (int e = 0; e < 4; ++e) { af[i].u[e] = p0[e]; af[i].u[4 + e] = p1[e]; }
    }
    // B frags: lane N=lane&15; halves K = half*8 + {0..7} and half*8+16+{0..7}
#pragma unroll
    for (int j = 0; j < 2; ++j) {
      const int n = waveN * 32 + j * 16 + row16;
      const uint32_t* p0 = reinterpret_cast<const uint32_t*>(&Bs[cur][n][half * 8]);
      const uint32_t* p1 = reinterpret_cast<const uint32_t*>(&Bs[cur][n][half * 8 + 16]);
#pragma unroll
      for (int e = 0; e < 4; ++e) { bfr[j].u[e] = p0[e]; bfr[j].u[4 + e] = p1[e]; }
    }
#pragma unroll
    for (int i = 0; i < 4; ++i)
#pragma unroll
      for (int j = 0; j < 2; ++j)
        acc[i][j] = wmma_bf16(af[i], bfr[j], acc[i][j]);

    // Drain next tile's loads into the other LDS buffer, then one barrier.
    if (kb + 1 < nk) lstore(cur ^ 1);
    __syncthreads();
  }

  // Epilogue. C layout: row = vgpr + half*8, col = lane&15.
#pragma unroll
  for (int i = 0; i < 4; ++i)
#pragma unroll
    for (int j = 0; j < 2; ++j) {
      const int r0 = bm0 + waveM * 64 + i * 16 + half * 8;
      const int c0 = bn0 + waveN * 32 + j * 16 + row16;
      const float bv = bias[c0];
#pragma unroll
      for (int r = 0; r < 8; ++r) {
        const size_t idx = (size_t)(r0 + r) * N + c0;
        float v = acc[i][j][r] + bv;
        if (epilogue == 0) {
          outF[idx] = v;
        } else if (epilogue == 1) {
          outH[idx] = f2bf(v);
        } else if (epilogue == 2) {
          v = 0.5f * v * (1.0f + erff(v * 0.70710678118654752f));
          outH[idx] = f2bf(v);
        } else {
          outF[idx] = residual[idx] + v;
        }
      }
    }
}

// ---------------------------------------------------------------------------
// Flash attention, one (b,h) per blockIdx.{z,y}; 4 waves x 16 Q-rows each.
// qkv: (B,S,3D) bf16; seg: (B,S) int; outO: (B,S,D) bf16.
// ---------------------------------------------------------------------------
__global__ __launch_bounds__(128) void flash_attn_wmma(
    const __bf16* __restrict__ qkv, const int* __restrict__ seg,
    __bf16* __restrict__ outO) {
  __shared__ __bf16 pbuf[4][16][32];   // per-wave P tile (C->A relayout)
  const int lane = threadIdx.x & 31;
  const int wave = threadIdx.x >> 5;
  const int half = lane >> 4;
  const int col = lane & 15;
  const int b = blockIdx.z;
  const int h = blockIdx.y;
  const int q0 = blockIdx.x * 64 + wave * 16;
  const size_t rstride = 3 * (size_t)kD;
  const __bf16* qbase = qkv + (size_t)b * kS * rstride + (size_t)h * kDH;
  const __bf16* kbase = qbase + kD;
  const __bf16* vbase = qbase + 2 * kD;

  // Q A-frags (16 rows x 64 d = 2 WMMA K-chunks), loaded straight from global
  Frag16 qa[2];
  {
    const __bf16* qr = qbase + (size_t)(q0 + col) * rstride;
#pragma unroll
    for (int f = 0; f < 2; ++f) {
      const uint32_t* p0 = reinterpret_cast<const uint32_t*>(qr + f * 32 + half * 8);
      const uint32_t* p1 = reinterpret_cast<const uint32_t*>(qr + f * 32 + half * 8 + 16);
#pragma unroll
      for (int e = 0; e < 4; ++e) { qa[f].u[e] = p0[e]; qa[f].u[4 + e] = p1[e]; }
    }
  }

  float mrow[8], ssum[8];
  int segq[8];
  f32x8 oacc[4];
#pragma unroll
  for (int r = 0; r < 8; ++r) {
    mrow[r] = -3.0e38f;
    ssum[r] = 0.0f;
    segq[r] = seg[b * kS + q0 + half * 8 + r];
  }
#pragma unroll
  for (int dt = 0; dt < 4; ++dt)
#pragma unroll
    for (int e = 0; e < 8; ++e) oacc[dt][e] = 0.0f;

  // Keys allowed: causal (k<=q) OR both in prefix; tiles beyond are skipped.
  const int kend = (q0 + 16 > kPrefix) ? (q0 + 16) : kPrefix;
  const int nkt = (kend + 31) / 32;
  for (int kt = 0; kt < nkt; ++kt) {
    const int kc0 = kt * 32;
    // Warm L2 with the next tile's K/V rows (one row per lane).
    if (kt + 1 < nkt) {
      const __bf16* nk_row = kbase + (size_t)(kc0 + 32 + lane) * rstride;
      __builtin_prefetch(nk_row, 0, 1);
      __builtin_prefetch(nk_row + kD, 0, 1);   // matching V row
    }
    // K^T B-frags: lane N=key-in-16; halves = K-row contiguous in d
    Frag16 kf[2][2];
#pragma unroll
    for (int ks = 0; ks < 2; ++ks) {
      const __bf16* kr = kbase + (size_t)(kc0 + ks * 16 + col) * rstride;
#pragma unroll
      for (int f = 0; f < 2; ++f) {
        const uint32_t* p0 = reinterpret_cast<const uint32_t*>(kr + f * 32 + half * 8);
        const uint32_t* p1 = reinterpret_cast<const uint32_t*>(kr + f * 32 + half * 8 + 16);
#pragma unroll
        for (int e = 0; e < 4; ++e) { kf[ks][f].u[e] = p0[e]; kf[ks][f].u[4 + e] = p1[e]; }
      }
    }
    f32x8 sc[2];
#pragma unroll
    for (int ks = 0; ks < 2; ++ks) {
#pragma unroll
      for (int e = 0; e < 8; ++e) sc[ks][e] = 0.0f;
      sc[ks] = wmma_bf16(qa[0], kf[ks][0], sc[ks]);
      sc[ks] = wmma_bf16(qa[1], kf[ks][1], sc[ks]);
    }

    const int segk0 = seg[b * kS + kc0 + col];
    const int segk1 = seg[b * kS + kc0 + 16 + col];
    float p0v[8], p1v[8];
#pragma unroll
    for (int r = 0; r < 8; ++r) {
      const int qpos = q0 + half * 8 + r;
      const int kp0 = kc0 + col;
      const int kp1 = kc0 + 16 + col;
      const bool a0 = ((kp0 <= qpos) || ((qpos < kPrefix) && (kp0 < kPrefix))) && (segk0 == segq[r]);
      const bool a1 = ((kp1 <= qpos) || ((qpos < kPrefix) && (kp1 < kPrefix))) && (segk1 == segq[r]);
      const float v0 = a0 ? sc[0][r] * 0.125f : -1.0e30f;   // 1/sqrt(64)
      const float v1 = a1 ? sc[1][r] * 0.125f : -1.0e30f;
      float tm = fmaxf(v0, v1);
      for (int m = 1; m < 16; m <<= 1) tm = fmaxf(tm, __shfl_xor(tm, m, 32));
      const float newm = fmaxf(mrow[r], tm);
      const float corr = __expf(mrow[r] - newm);
      const float e0 = a0 ? __expf(v0 - newm) : 0.0f;
      const float e1 = a1 ? __expf(v1 - newm) : 0.0f;
      float rs = e0 + e1;
      for (int m = 1; m < 16; m <<= 1) rs += __shfl_xor(rs, m, 32);
      ssum[r] = ssum[r] * corr + rs;
      mrow[r] = newm;
#pragma unroll
      for (int dt = 0; dt < 4; ++dt) oacc[dt][r] *= corr;
      p0v[r] = e0; p1v[r] = e1;
    }

    // P (C layout) -> LDS -> A-frag layout, per-wave region, DS-cnt fenced
#pragma unroll
    for (int r = 0; r < 8; ++r) {
      pbuf[wave][half * 8 + r][col]      = f2bf(p0v[r]);
      pbuf[wave][half * 8 + r][16 + col] = f2bf(p1v[r]);
    }
    asm volatile("s_wait_dscnt 0" ::: "memory");
    Frag16 pa;
    {
      const uint32_t* p0 = reinterpret_cast<const uint32_t*>(&pbuf[wave][col][half * 8]);
      const uint32_t* p1 = reinterpret_cast<const uint32_t*>(&pbuf[wave][col][half * 8 + 16]);
#pragma unroll
      for (int e = 0; e < 4; ++e) { pa.u[e] = p0[e]; pa.u[4 + e] = p1[e]; }
    }

    // V B-frags (32 keys x 64 d = 4 N-tiles); gather per-lane key rows.
    // Lanes of a half-wave read consecutive d for the same key row, so the
    // 16-bit gathers coalesce into 32B segments per row.
#pragma unroll
    for (int dt = 0; dt < 4; ++dt) {
      Frag16 vf;
      const __bf16* vc = vbase + dt * 16 + col;
#pragma unroll
      for (int j = 0; j < 8; ++j) {
        const int key = kc0 + half * 8 + 2 * j + (j >= 4 ? 8 : 0);
        vf.h[2 * j]     = vc[(size_t)key * rstride];
        vf.h[2 * j + 1] = vc[(size_t)(key + 1) * rstride];
      }
      oacc[dt] = wmma_bf16(pa, vf, oacc[dt]);
    }
  }

  // Normalize and emit O as bf16 (A input of the out-proj GEMM)
#pragma unroll
  for (int dt = 0; dt < 4; ++dt)
#pragma unroll
    for (int r = 0; r < 8; ++r) {
      const int qpos = q0 + half * 8 + r;
      const float v = oacc[dt][r] / ssum[r];
      outO[(size_t)b * kS * kD + (size_t)qpos * kD + h * kDH + dt * 16 + col] = f2bf(v);
    }
}

// ---------------------------------------------------------------------------
extern "C" void kernel_launch(void* const* d_in, const int* in_sizes, int n_in,
                              void* d_out, int out_size, void* d_ws, size_t ws_size,
                              hipStream_t stream) {
  (void)in_sizes; (void)n_in; (void)out_size; (void)ws_size;
  const int*   input_ids   = (const int*)d_in[0];
  const int*   segment_ids = (const int*)d_in[1];
  // d_in[2] attention_mask == (arange(S) < PREFIX): folded into kernel constant
  const float* embedding   = (const float*)d_in[3];
  const float* in_proj_w   = (const float*)d_in[4];
  const float* in_proj_b   = (const float*)d_in[5];
  const float* out_proj_w  = (const float*)d_in[6];
  const float* out_proj_b  = (const float*)d_in[7];
  const float* ln1_g       = (const float*)d_in[8];
  const float* ln1_b       = (const float*)d_in[9];
  const float* ln2_g       = (const float*)d_in[10];
  const float* ln2_b       = (const float*)d_in[11];
  const float* lin1_w      = (const float*)d_in[12];
  const float* lin1_b      = (const float*)d_in[13];
  const float* lin2_w      = (const float*)d_in[14];
  const float* lin2_b      = (const float*)d_in[15];
  const float* normf_g     = (const float*)d_in[16];
  const float* normf_b     = (const float*)d_in[17];
  const float* head_w      = (const float*)d_in[18];
  const float* head_b      = (const float*)d_in[19];
  float* logits = (float*)d_out;

  // Workspace carve-out (~44 MB), everything written before read.
  char* w = (char*)d_ws;
  float*  x     = (float*)w;  w += (size_t)kRows * kD * sizeof(float);
  __bf16* hbf   = (__bf16*)w; w += (size_t)kRows * kD * sizeof(__bf16);
  __bf16* qkvbf = (__bf16*)w; w += (size_t)kRows * 3 * kD * sizeof(__bf16);
  __bf16* obf   = (__bf16*)w; w += (size_t)kRows * kD * sizeof(__bf16);
  __bf16* ffbf  = (__bf16*)w; w += (size_t)kRows * kFF * sizeof(__bf16);

  embed_gather<<<kRows, 256, 0, stream>>>(input_ids, embedding, x);

  for (int l = 0; l < kL; ++l) {
    layernorm_to_bf16<<<kRows, 256, 0, stream>>>(
        x, ln1_g + (size_t)l * kD, ln1_b + (size_t)l * kD, hbf);
    gemm_bf16_wmma<<<dim3(3 * kD / GBN, kRows / GBM), 256, 0, stream>>>(
        hbf, in_proj_w + (size_t)l * 3 * kD * kD, in_proj_b + (size_t)l * 3 * kD,
        nullptr, nullptr, qkvbf, kRows, 3 * kD, kD, /*epilogue=*/1);
    flash_attn_wmma<<<dim3(kS / 64, kH, kB), 128, 0, stream>>>(
        qkvbf, segment_ids, obf);
    gemm_bf16_wmma<<<dim3(kD / GBN, kRows / GBM), 256, 0, stream>>>(
        obf, out_proj_w + (size_t)l * kD * kD, out_proj_b + (size_t)l * kD,
        x, x, nullptr, kRows, kD, kD, /*epilogue=*/3);
    layernorm_to_bf16<<<kRows, 256, 0, stream>>>(
        x, ln2_g + (size_t)l * kD, ln2_b + (size_t)l * kD, hbf);
    gemm_bf16_wmma<<<dim3(kFF / GBN, kRows / GBM), 256, 0, stream>>>(
        hbf, lin1_w + (size_t)l * kFF * kD, lin1_b + (size_t)l * kFF,
        nullptr, nullptr, ffbf, kRows, kFF, kD, /*epilogue=*/2);
    gemm_bf16_wmma<<<dim3(kD / GBN, kRows / GBM), 256, 0, stream>>>(
        ffbf, lin2_w + (size_t)l * kD * kFF, lin2_b + (size_t)l * kD,
        x, x, nullptr, kRows, kD, kFF, /*epilogue=*/3);
  }

  layernorm_to_bf16<<<kRows, 256, 0, stream>>>(x, normf_g, normf_b, hbf);
  gemm_bf16_wmma<<<dim3(kV / GBN, kRows / GBM), 256, 0, stream>>>(
      hbf, head_w, head_b, nullptr, logits, nullptr, kRows, kV, kD, /*epilogue=*/0);
}
